// MambaModel_41815801594537
// MI455X (gfx1250) — matrix operand order repaired and checked by reading, not compiled, not run on hardware
//
#include <hip/hip_runtime.h>
#include <hip/hip_bf16.h>

typedef __attribute__((ext_vector_type(16))) _Float16 v16h;
typedef __attribute__((ext_vector_type(8)))  _Float16 v8h;
typedef __attribute__((ext_vector_type(8)))  float    v8f;

// Problem constants (match reference)
constexpr int B_  = 8;
constexpr int L_  = 2048;
constexpr int D_  = 768;
constexpr int E_  = 1536;   // 2*D
constexpr int S_  = 16;
constexpr int K_  = 4;
constexpr int R_  = 48;
constexpr int NL_ = 4;
constexpr int M_  = B_ * L_;            // 16384 rows
constexpr float EPS_ = 1e-5f;

// LDS tile geometry for the WMMA GEMM
constexpr int BM_ = 128;                // block tile M
constexpr int BN_ = 128;                // block tile N
constexpr int BK_ = 32;                 // K step (one WMMA K)
constexpr int LDS_ROW_ = 40;            // 32 halves + 8 pad (80B row stride, spreads banks)

// -------------------------------------------------------------------------
// Block-wide float sum reduction (wave32-aware)
// -------------------------------------------------------------------------
__device__ __forceinline__ float block_reduce_sum(float v) {
    __shared__ float red[8];
    __syncthreads();                      // protect shared reuse across calls
    int lane = threadIdx.x & 31;
    int wave = threadIdx.x >> 5;
#pragma unroll
    for (int o = 16; o > 0; o >>= 1) v += __shfl_xor(v, o, 32);
    if (lane == 0) red[wave] = v;
    __syncthreads();
    int nw = blockDim.x >> 5;
    float t = (threadIdx.x < nw) ? red[threadIdx.x] : 0.f;
    if (wave == 0) {
#pragma unroll
        for (int o = 4; o > 0; o >>= 1) t += __shfl_xor(t, o, 32);
        if (lane == 0) red[0] = t;
    }
    __syncthreads();
    return red[0];
}

// -------------------------------------------------------------------------
// (B, D, L) -> (B, L, D) tiled transpose
// -------------------------------------------------------------------------
__global__ void transpose_bdl_kernel(const float* __restrict__ in, float* __restrict__ out) {
    __shared__ float tile[32][33];
    int b  = blockIdx.z;
    int l0 = blockIdx.x * 32;
    int d0 = blockIdx.y * 32;
    for (int i = threadIdx.y; i < 32; i += 8)
        tile[i][threadIdx.x] = in[((size_t)b * D_ + d0 + i) * L_ + l0 + threadIdx.x];
    __syncthreads();
    for (int i = threadIdx.y; i < 32; i += 8)
        out[((size_t)b * L_ + l0 + i) * D_ + d0 + threadIdx.x] = tile[threadIdx.x][i];
}

// -------------------------------------------------------------------------
// Row-wise RMSNorm -> f16 output (one block per row)
// -------------------------------------------------------------------------
__global__ void rmsnorm_f16_kernel(const float* __restrict__ x, const float* __restrict__ w,
                                   _Float16* __restrict__ out, int Dd) {
    size_t row = blockIdx.x;
    const float* xr = x + row * Dd;
    float ss = 0.f;
    for (int i = threadIdx.x; i < Dd; i += blockDim.x) { float v = xr[i]; ss += v * v; }
    ss = block_reduce_sum(ss);
    float sc = rsqrtf(ss / (float)Dd + EPS_);
    for (int i = threadIdx.x; i < Dd; i += blockDim.x)
        out[row * Dd + i] = (_Float16)(xr[i] * sc * w[i]);
}

// -------------------------------------------------------------------------
// fp32 -> f16 conversion (weights)
// -------------------------------------------------------------------------
__global__ void f32_to_f16_kernel(const float* __restrict__ in, _Float16* __restrict__ out, size_t n) {
    size_t i = (size_t)blockIdx.x * blockDim.x + threadIdx.x;
    if (i < n) out[i] = (_Float16)in[i];
}

// dt_proj_w (E,48) -> f16 (E,64) with zero pad in cols 48..63
__global__ void dtw_pad_kernel(const float* __restrict__ in, _Float16* __restrict__ out) {
    size_t i = (size_t)blockIdx.x * blockDim.x + threadIdx.x;
    if (i >= (size_t)E_ * 64) return;
    int col = (int)(i & 63);
    int row = (int)(i >> 6);
    out[i] = (col < R_) ? (_Float16)in[(size_t)row * R_ + col] : (_Float16)0.f;
}

// -------------------------------------------------------------------------
// WMMA fragment loads from an LDS tile (16x16x32 f16 layouts).
// A (16x32, MxK): lane holds row m0+(lane&15); K halves {hi*8..+7, 16+hi*8..+7}
// B (32x16, KxN): lane holds W row n0+(lane&15) (= column of B); K halves {hi*8..+7, 16+hi*8..+7}
// -------------------------------------------------------------------------
__device__ __forceinline__ v16h lds_fragA(const _Float16* s, int m0, int lane) {
    const int r = lane & 15, hi = lane >> 4;
    const _Float16* p = s + (m0 + r) * LDS_ROW_ + hi * 8;
    v8h lo = *(const v8h*)(p);
    v8h hv = *(const v8h*)(p + 16);
    v16h f;
#pragma unroll
    for (int i = 0; i < 8; ++i) { f[i] = lo[i]; f[i + 8] = hv[i]; }
    return f;
}

__device__ __forceinline__ v16h lds_fragB(const _Float16* s, int n0, int lane) {
    const int r = lane & 15, hi = lane >> 4;
    const _Float16* p = s + (n0 + r) * LDS_ROW_ + hi * 8;
    v8h lo = *(const v8h*)(p);
    v8h hv = *(const v8h*)(p + 16);
    v16h f;
#pragma unroll
    for (int i = 0; i < 8; ++i) { f[i] = lo[i]; f[i + 8] = hv[i]; }
    return f;
}

// -------------------------------------------------------------------------
// LDS-staged, double-buffered WMMA GEMM: out = epilogue(A(MxK) * W(NxK)^T)
//   mode 0: store fp32 (outF) and/or f16 (outH)
//   mode 1: softplus(x + bias[col]) -> outF
//   mode 2: x + resid[row*ldo+col]  -> outF
// Block: 256 threads = 8 waves (4 along M x 2 along N); block tile 128x128.
// Wave tile: 32x64 = 2x4 WMMA tiles (8 v_wmma per K step).
// Rows of W beyond N are zero-filled in LDS; control flow is barrier-uniform.
// Requires: M % 128 == 0, K % 32 == 0, N % 16 == 0 (all hold here).
// -------------------------------------------------------------------------
__global__ __launch_bounds__(256) void wmma_gemm_kernel(
    const _Float16* __restrict__ A, const _Float16* __restrict__ W,
    int M, int N, int K, int lda, int ldw,
    float* __restrict__ outF, _Float16* __restrict__ outH, int ldo,
    const float* __restrict__ bias, const float* __restrict__ resid, int mode) {

    __shared__ _Float16 sA[2][BM_ * LDS_ROW_];
    __shared__ _Float16 sB[2][BN_ * LDS_ROW_];

    const int tid  = threadIdx.x;
    const int lane = tid & 31;
    const int wave = tid >> 5;
    const int wm   = wave & 3;          // 0..3 along M
    const int wn   = wave >> 2;         // 0..1 along N

    const int m_blk = blockIdx.y * BM_;
    const int n_blk = blockIdx.x * BN_;

    // cooperative-fill mapping: thread -> (row, 8-half segment), 2 rows per thread
    const int lrow = tid >> 2;          // 0..63
    const int lseg = tid & 3;           // 0..3 (16B chunk within 32-half row)

    v8f acc[2][4];
#pragma unroll
    for (int mi = 0; mi < 2; ++mi)
#pragma unroll
        for (int ni = 0; ni < 4; ++ni) acc[mi][ni] = {};

    v8h ra[2], rb[2];
    const v8h zz = {};
    auto gload = [&](int k0) {
#pragma unroll
        for (int p = 0; p < 2; ++p) {
            int row = lrow + p * 64;
            ra[p] = *(const v8h*)(A + (size_t)(m_blk + row) * lda + k0 + lseg * 8);
            int n = n_blk + row;
            rb[p] = (n < N) ? *(const v8h*)(W + (size_t)n * ldw + k0 + lseg * 8) : zz;
        }
    };
    auto sstore = [&](int buf) {
#pragma unroll
        for (int p = 0; p < 2; ++p) {
            int row = lrow + p * 64;
            *(v8h*)(&sA[buf][row * LDS_ROW_ + lseg * 8]) = ra[p];
            *(v8h*)(&sB[buf][row * LDS_ROW_ + lseg * 8]) = rb[p];
        }
    };

    const int nk = K / BK_;
    gload(0);
    sstore(0);
    __syncthreads();

    for (int k = 0; k < nk; ++k) {
        if (k + 1 < nk) gload((k + 1) * BK_);

        const _Float16* bA = sA[k & 1];
        const _Float16* bB = sB[k & 1];
        v16h a0 = lds_fragA(bA, wm * 32,      lane);
        v16h a1 = lds_fragA(bA, wm * 32 + 16, lane);
#pragma unroll
        for (int ni = 0; ni < 4; ++ni) {
            v16h bf = lds_fragB(bB, wn * 64 + ni * 16, lane);
            acc[0][ni] = __builtin_amdgcn_wmma_f32_16x16x32_f16(false, a0, false, bf, (short)0, acc[0][ni], false, false);
            acc[1][ni] = __builtin_amdgcn_wmma_f32_16x16x32_f16(false, a1, false, bf, (short)0, acc[1][ni], false, false);
        }

        if (k + 1 < nk) sstore((k + 1) & 1);
        __syncthreads();
    }

    // epilogue
    const int r  = lane & 15;
    const int hi = lane >> 4;
#pragma unroll
    for (int mi = 0; mi < 2; ++mi) {
#pragma unroll
        for (int ni = 0; ni < 4; ++ni) {
            int nt = n_blk + wn * 64 + ni * 16;
            if (nt >= N) continue;
            int mt = m_blk + wm * 32 + mi * 16;
#pragma unroll
            for (int j = 0; j < 8; ++j) {
                int row = mt + hi * 8 + j;
                int col = nt + r;
                float v = acc[mi][ni][j];
                if (mode == 1) { v += bias[col]; v = (v > 20.f) ? v : log1pf(expf(v)); }
                else if (mode == 2) { v += resid[(size_t)row * ldo + col]; }
                size_t o = (size_t)row * ldo + col;
                if (outF) outF[o] = v;
                if (outH) outH[o] = (_Float16)v;
            }
        }
    }
}

// -------------------------------------------------------------------------
// Causal depthwise conv (K=4) + SiLU.  xz:(B,L,2E) -> xc fp32 + f16 (B,L,E)
// -------------------------------------------------------------------------
__global__ void conv_silu_kernel(const float* __restrict__ xz,
                                 const float* __restrict__ cw, const float* __restrict__ cb,
                                 float* __restrict__ xc, _Float16* __restrict__ xc16) {
    size_t idx = (size_t)blockIdx.x * blockDim.x + threadIdx.x;
    if (idx >= (size_t)M_ * E_) return;
    int e = (int)(idx % E_);
    size_t bl = idx / E_;
    int t = (int)(bl % L_);
    int b = (int)(bl / L_);
    const float* base = xz + (size_t)b * L_ * (2 * E_) + e;
    float acc = cb[e];
#pragma unroll
    for (int k = 0; k < K_; ++k) {
        int tt = t - (K_ - 1) + k;
        if (tt >= 0) acc += cw[e * K_ + k] * base[(size_t)tt * (2 * E_)];
    }
    float s = acc / (1.f + __expf(-acc));   // silu
    xc[idx] = s;
    xc16[idx] = (_Float16)s;
}

// -------------------------------------------------------------------------
// Selective scan, fused with +u*D, *silu(z), f16 output for out_proj.
// grid (B_, E_/256); thread = one channel, 16 states in registers.
// -------------------------------------------------------------------------
__global__ __launch_bounds__(256) void scan_kernel(
    const float* __restrict__ xc, const float* __restrict__ dt,
    const float* __restrict__ dbl, const float* __restrict__ alog,
    const float* __restrict__ Dp, const float* __restrict__ xz,
    _Float16* __restrict__ yz16) {

    int b = blockIdx.x;
    int e = blockIdx.y * blockDim.x + threadIdx.x;

    float A[S_], h[S_];
#pragma unroll
    for (int s = 0; s < S_; ++s) { A[s] = -__expf(alog[(size_t)e * S_ + s]); h[s] = 0.f; }
    float De = Dp[e];

    __shared__ float sB[S_], sC[S_];
    const size_t rowbase = (size_t)b * L_;

    for (int t = 0; t < L_; ++t) {
        size_t r = rowbase + t;
        if (threadIdx.x < 32) {
            float v = dbl[r * 80 + R_ + threadIdx.x];
            if (threadIdx.x < S_) sB[threadIdx.x] = v;
            else                  sC[threadIdx.x - S_] = v;
        }
        __syncthreads();
        float dtv = dt[r * E_ + e];
        float u   = xc[r * E_ + e];
        float du  = dtv * u;
        float y   = 0.f;
#pragma unroll
        for (int s = 0; s < S_; ++s) {
            float dA = __expf(dtv * A[s]);
            h[s] = dA * h[s] + du * sB[s];
            y += h[s] * sC[s];
        }
        y += u * De;
        float z = xz[r * (2 * E_) + E_ + e];
        float g = y * (z / (1.f + __expf(-z)));
        yz16[r * E_ + e] = (_Float16)g;
        __syncthreads();
    }
}

// -------------------------------------------------------------------------
// Final: idx = popcount(mask[b]) - 1; rmsnorm(x[b, idx]) * normf_w -> out
// -------------------------------------------------------------------------
__global__ void final_kernel(const float* __restrict__ x, const unsigned char* __restrict__ mask,
                             const float* __restrict__ w, float* __restrict__ out) {
    int b = blockIdx.x;
    float cnt = 0.f;
    for (int i = threadIdx.x; i < L_; i += blockDim.x)
        cnt += mask[(size_t)b * L_ + i] ? 1.f : 0.f;
    float tot = block_reduce_sum(cnt);
    int idx = (int)(tot + 0.5f) - 1;
    if (idx < 0) idx = 0;
    const float* row = x + ((size_t)b * L_ + idx) * D_;
    float ss = 0.f;
    for (int i = threadIdx.x; i < D_; i += blockDim.x) { float v = row[i]; ss += v * v; }
    ss = block_reduce_sum(ss);
    float sc = rsqrtf(ss / (float)D_ + EPS_);
    for (int i = threadIdx.x; i < D_; i += blockDim.x)
        out[(size_t)b * D_ + i] = row[i] * sc * w[i];
}

// -------------------------------------------------------------------------
// Host orchestration
// -------------------------------------------------------------------------
extern "C" void kernel_launch(void* const* d_in, const int* in_sizes, int n_in,
                              void* d_out, int out_size, void* d_ws, size_t ws_size,
                              hipStream_t stream) {
    (void)in_sizes; (void)n_in; (void)out_size; (void)ws_size;

    const float* batch   = (const float*)d_in[0];
    const unsigned char* mask = (const unsigned char*)d_in[1];
    const float* norm_w  = (const float*)d_in[2];
    const float* ipw     = (const float*)d_in[3];   // (NL, 2E, D)
    const float* cw      = (const float*)d_in[4];   // (NL, E, K)
    const float* cb      = (const float*)d_in[5];   // (NL, E)
    const float* xpw     = (const float*)d_in[6];   // (NL, 80, E)
    const float* dtw     = (const float*)d_in[7];   // (NL, E, R)
    const float* dtb     = (const float*)d_in[8];   // (NL, E)
    const float* alog    = (const float*)d_in[9];   // (NL, E, S)
    const float* Dp      = (const float*)d_in[10];  // (NL, E)
    const float* opw     = (const float*)d_in[11];  // (NL, D, E)
    const float* normf_w = (const float*)d_in[12];  // (D)
    float* out = (float*)d_out;

    char* ws = (char*)d_ws;
    size_t off = 0;
    auto alloc = [&](size_t bytes) -> char* {
        char* p = ws + off;
        off += (bytes + 255) & ~(size_t)255;
        return p;
    };
    float*    X0    = (float*)alloc((size_t)M_ * D_ * 4);
    float*    X1    = (float*)alloc((size_t)M_ * D_ * 4);
    _Float16* H16   = (_Float16*)alloc((size_t)M_ * D_ * 2);
    float*    XZ    = (float*)alloc((size_t)M_ * 2 * E_ * 4);
    float*    XC    = (float*)alloc((size_t)M_ * E_ * 4);
    _Float16* XC16  = (_Float16*)alloc((size_t)M_ * E_ * 2);
    float*    DBL   = (float*)alloc((size_t)M_ * 80 * 4);
    _Float16* DBL16 = (_Float16*)alloc((size_t)M_ * 80 * 2);
    float*    DT    = (float*)alloc((size_t)M_ * E_ * 4);
    _Float16* YZ16  = (_Float16*)alloc((size_t)M_ * E_ * 2);
    _Float16* WIP16 = (_Float16*)alloc((size_t)2 * E_ * D_ * 2);
    _Float16* WXP16 = (_Float16*)alloc((size_t)80 * E_ * 2);
    _Float16* WDT16 = (_Float16*)alloc((size_t)E_ * 64 * 2);
    _Float16* WOP16 = (_Float16*)alloc((size_t)D_ * E_ * 2);

    // batch (B,D,L) -> X0 (B,L,D)
    transpose_bdl_kernel<<<dim3(L_ / 32, D_ / 32, B_), dim3(32, 8), 0, stream>>>(batch, X0);

    float* xcur = X0;
    float* xnext = X1;

    auto gemm = [&](const _Float16* A, int lda, const _Float16* W, int ldw,
                    int N, int K, float* oF, _Float16* oH, int ldo,
                    const float* bias, const float* resid, int mode) {
        dim3 grid((N + BN_ - 1) / BN_, (M_ + BM_ - 1) / BM_);
        wmma_gemm_kernel<<<grid, 256, 0, stream>>>(A, W, M_, N, K, lda, ldw,
                                                   oF, oH, ldo, bias, resid, mode);
    };
    auto cvt = [&](const float* src, _Float16* dst, size_t n) {
        f32_to_f16_kernel<<<(unsigned)((n + 255) / 256), 256, 0, stream>>>(src, dst, n);
    };

    for (int i = 0; i < NL_; ++i) {
        // weight conversions for this layer
        cvt(ipw + (size_t)i * 2 * E_ * D_, WIP16, (size_t)2 * E_ * D_);
        cvt(xpw + (size_t)i * 80 * E_,     WXP16, (size_t)80 * E_);
        cvt(opw + (size_t)i * D_ * E_,     WOP16, (size_t)D_ * E_);
        dtw_pad_kernel<<<(E_ * 64 + 255) / 256, 256, 0, stream>>>(dtw + (size_t)i * E_ * R_, WDT16);

        // h = rmsnorm(x) -> f16
        rmsnorm_f16_kernel<<<M_, 256, 0, stream>>>(xcur, norm_w + (size_t)i * D_, H16, D_);

        // xz = h @ in_proj_w^T   (M x 3072)
        gemm(H16, D_, WIP16, D_, 2 * E_, D_, XZ, nullptr, 2 * E_, nullptr, nullptr, 0);

        // xc = silu(causal_dw_conv(xz[..., :E]))
        conv_silu_kernel<<<(unsigned)(((size_t)M_ * E_ + 255) / 256), 256, 0, stream>>>(
            XZ, cw + (size_t)i * E_ * K_, cb + (size_t)i * E_, XC, XC16);

        // dbl = xc @ x_proj_w^T  (M x 80), keep both fp32 and f16
        gemm(XC16, E_, WXP16, E_, 80, E_, DBL, DBL16, 80, nullptr, nullptr, 0);

        // dt = softplus(dbl[:, :48] @ dt_proj_w^T + dtb)   (K padded to 64)
        gemm(DBL16, 80, WDT16, 64, E_, 64, DT, nullptr, E_, dtb + (size_t)i * E_, nullptr, 1);

        // selective scan fused with  (+u*D) * silu(z)  -> f16
        scan_kernel<<<dim3(B_, E_ / 256), 256, 0, stream>>>(
            XC, DT, DBL, alog + (size_t)i * E_ * S_, Dp + (size_t)i * E_, XZ, YZ16);

        // x = x + yz @ out_proj_w^T
        gemm(YZ16, E_, WOP16, E_, D_, E_, xnext, nullptr, D_, nullptr, xcur, 2);

        float* tmp = xcur; xcur = xnext; xnext = tmp;
    }

    final_kernel<<<B_, 256, 0, stream>>>(xcur, mask, normf_w, out);
}
